// STaRC_12824772346471
// MI455X (gfx1250) — matrix operand order repaired and checked by reading, not compiled, not run on hardware
//
#include <hip/hip_runtime.h>
#include <math.h>

// ---------------------------------------------------------------------------
// Multiscale window attention (w = 8,32,64) fused into one CDNA5 kernel.
//   per 64-token tile:  S = X X^T / sqrt(D)        (WMMA f16 -> f32)
//                       P = sum of 3 nested softmaxes of S, / (3+eps)
//                       O = P X                    (WMMA f16 -> f32)
//                       out = video + LayerNorm(O) * gamma + beta
// B=32, T=1024, D=768  ->  512 workgroups of 256 threads (8 wave32).
//
// LDS: Xh [64][776] f16 row-major copy (reused in-place for O)
//      Xt [768][72] f16 transposed copy (kills the B-fragment gather)
//      S  [64][68]  f32 scores, P [64][72] f16 fused attention weights
// ---------------------------------------------------------------------------

typedef __attribute__((ext_vector_type(16))) _Float16 v16h;
typedef __attribute__((ext_vector_type(8)))  _Float16 v8h;
typedef __attribute__((ext_vector_type(4)))  _Float16 v4h;
typedef __attribute__((ext_vector_type(8)))  float    v8f;

#define TILE_T 64
#define DDIM   768
#define XS     776   // Xh row stride in halves (1552B, 16B aligned)
#define TS     72    // Xt row stride in halves (144B, 16B aligned)
#define SS     68    // S row stride in floats
#define PS     72    // P row stride in halves (144B, 16B aligned)

#define SMEM_BYTES ((size_t)TILE_T*XS*2 + (size_t)DDIM*TS*2 + \
                    (size_t)TILE_T*SS*4 + (size_t)TILE_T*PS*2)

static __device__ __forceinline__ v16h cat8(v8h lo, v8h hi) {
  return __builtin_shufflevector(lo, hi, 0,1,2,3,4,5,6,7,8,9,10,11,12,13,14,15);
}

__global__ __launch_bounds__(256) void msw_attn_ln_kernel(
    const float* __restrict__ video,
    const float* __restrict__ ln_gamma,
    const float* __restrict__ ln_beta,
    float* __restrict__ out)
{
  extern __shared__ char smem[];
  _Float16* Xh = (_Float16*)smem;                      // [64][XS]
  _Float16* Xt = Xh + (size_t)TILE_T * XS;             // [768][TS]
  float*    S  = (float*)(Xt + (size_t)DDIM * TS);     // [64][SS]
  _Float16* P  = (_Float16*)(S + (size_t)TILE_T * SS); // [64][PS]

  const int tid  = threadIdx.x;
  const int lane = tid & 31;
  const int wave = tid >> 5;

  const int blk = blockIdx.x;
  const int bb  = blk >> 4;              // batch index (16 tiles per batch)
  const int t0  = (blk & 15) * TILE_T;   // token base inside T=1024
  const size_t gbase = ((size_t)bb * 1024 + t0) * DDIM;

  // ---------------- Phase 1: global f32 -> LDS f16 (row-major + transposed)
  {
    const float4* src = (const float4*)(video + gbase);
    for (int idx = tid; idx < TILE_T * (DDIM / 4); idx += 256) {
      const int r  = idx / (DDIM / 4);
      const int c4 = idx % (DDIM / 4);
      float4 v = src[r * (DDIM / 4) + c4];
      v4h h = { (_Float16)v.x, (_Float16)v.y, (_Float16)v.z, (_Float16)v.w };
      *(v4h*)(Xh + r * XS + c4 * 4) = h;
      const int d = c4 * 4;
      Xt[(d + 0) * TS + r] = h[0];
      Xt[(d + 1) * TS + r] = h[1];
      Xt[(d + 2) * TS + r] = h[2];
      Xt[(d + 3) * TS + r] = h[3];
    }
  }
  __syncthreads();

  // ---------------- Phase 2: S = Xh * Xh^T / sqrt(D) -----------------------
  // 16 score tiles (4x4 of 16x16); each wave owns tiles {wave, wave+8}.
  // A from Xh (row-major), B also from Xh (B[k][n] = Xh[n][k]) -> all ds_load_b128.
  {
    const float rscale = 0.03608439182435161f;   // 1/sqrt(768)
    const int   kb     = (lane < 16) ? 0 : 8;    // A-frag K base (ISA layout)
    const int   am     = lane & 15;              // A-frag M = lane&15
    for (int t = wave; t < 16; t += 8) {
      const int mi = t >> 2, ni = t & 3;
      const int arow = (mi * 16 + am) * XS;
      const int bcol = ni * 16;
      v8f acc = {};
      for (int kk = 0; kk < DDIM / 32; ++kk) {
        const int k0 = kk * 32;
        // A: 16x32 f16 fragment, two aligned 16B LDS loads per lane
        v8h alo = *(const v8h*)(Xh + arow + k0 + kb);
        v8h ahi = *(const v8h*)(Xh + arow + k0 + kb + 16);
        // B: B[k][n] = X[n][k] = Xh[n][k]; lane = n, contiguous k per lane (ISA layout)
        const int brow = (bcol + am) * XS + k0 + kb;
        v8h blo = *(const v8h*)(Xh + brow);
        v8h bhi = *(const v8h*)(Xh + brow + 16);
        acc = __builtin_amdgcn_wmma_f32_16x16x32_f16(
                  false, cat8(alo, ahi), false, cat8(blo, bhi),
                  (short)0, acc, false, false);
      }
      const int moff = (lane < 16) ? 0 : 8;
      const int ncol = lane & 15;
      #pragma unroll
      for (int r = 0; r < 8; ++r)
        S[(mi * 16 + moff + r) * SS + ni * 16 + ncol] = acc[r] * rscale;
    }
  }
  __syncthreads();

  // ---------------- Phase 3: fused 3-scale softmax -> P --------------------
  // One thread per row. The diagonal term sits in every nested window, so a
  // single global row max is numerically safe for all three softmaxes.
  if (tid < TILE_T) {
    const int i = tid;
    float* srow = S + i * SS;
    float m = -1e30f;
    #pragma unroll
    for (int j = 0; j < 64; ++j) m = fmaxf(m, srow[j]);
    const int b32 = i & 32;          // 32-window base within the tile
    const int b8  = i & 56;          // 8-window base within the tile
    float s64 = 0.f, s32 = 0.f, s8 = 0.f;
    #pragma unroll
    for (int j = 0; j < 64; ++j) {
      const float e = __expf(srow[j] - m);
      srow[j] = e;                   // stream exp back through LDS (no big array)
      s64 += e;
      if ((j & 32) == b32) s32 += e;
      if ((j & 56) == b8)  s8  += e;
    }
    const float i64 = 1.f / s64, i32v = 1.f / s32, i8v = 1.f / s8;
    const float inv_cnt = 1.0f / (3.0f + 1e-6f);   // acc / (cnt + CNT_EPS)
    #pragma unroll
    for (int j = 0; j < 64; ++j) {
      const float e = srow[j];
      float p = e * i64;
      if ((j & 32) == b32) p += e * i32v;
      if ((j & 56) == b8)  p += e * i8v;
      P[i * PS + j] = (_Float16)(p * inv_cnt);
    }
  }
  __syncthreads();

  // ---------------- Phase 4: O = P * Xh, written in place over Xh ----------
  // 48 D-slices of 16 columns; one wave per slice => safe in-place overwrite
  // (B fragments pulled into registers before any store).
  {
    const int kb = (lane < 16) ? 0 : 8;
    const int am = lane & 15;
    for (int s = wave; s < DDIM / 16; s += 8) {
      const int d0 = s * 16;
      // B: Xt[d0+n][k_token], lane = n, contiguous k per lane (ISA layout)
      v8h lo = *(const v8h*)(Xt + (d0 + am) * TS + kb);
      v8h hi = *(const v8h*)(Xt + (d0 + am) * TS + kb + 16);
      const v16h b0 = cat8(lo, hi);
      lo = *(const v8h*)(Xt + (d0 + am) * TS + 32 + kb);
      hi = *(const v8h*)(Xt + (d0 + am) * TS + 32 + kb + 16);
      const v16h b1 = cat8(lo, hi);

      v8f o[4];
      #pragma unroll
      for (int mt = 0; mt < 4; ++mt) {
        const int arow = (mt * 16 + am) * PS;
        v8h a0l = *(const v8h*)(P + arow + kb);
        v8h a0h = *(const v8h*)(P + arow + kb + 16);
        v8h a1l = *(const v8h*)(P + arow + 32 + kb);
        v8h a1h = *(const v8h*)(P + arow + 32 + kb + 16);
        v8f acc = {};
        acc = __builtin_amdgcn_wmma_f32_16x16x32_f16(
                  false, cat8(a0l, a0h), false, b0, (short)0, acc, false, false);
        acc = __builtin_amdgcn_wmma_f32_16x16x32_f16(
                  false, cat8(a1l, a1h), false, b1, (short)0, acc, false, false);
        o[mt] = acc;
      }
      const int moff = (lane < 16) ? 0 : 8;
      const int ncol = lane & 15;
      #pragma unroll
      for (int mt = 0; mt < 4; ++mt) {
        #pragma unroll
        for (int r = 0; r < 8; ++r)
          Xh[(mt * 16 + moff + r) * XS + d0 + ncol] = (_Float16)o[mt][r];
      }
    }
  }
  __syncthreads();

  // ---------------- Phase 5: LayerNorm + residual + store (vectorized) -----
  // Xh now holds vz = acc/(3+eps) in f16. 8 rows per wave; wave32 reduction.
  for (int r = wave * 8; r < wave * 8 + 8; ++r) {
    float sum = 0.f, sq = 0.f;
    float vbuf[24];
    #pragma unroll
    for (int i = 0; i < 6; ++i) {
      v4h h = *(const v4h*)(Xh + r * XS + lane * 4 + i * 128);
      #pragma unroll
      for (int j = 0; j < 4; ++j) {
        const float v = (float)h[j];
        vbuf[i * 4 + j] = v;
        sum += v; sq += v * v;
      }
    }
    #pragma unroll
    for (int off = 16; off > 0; off >>= 1) {
      sum += __shfl_xor(sum, off, 32);
      sq  += __shfl_xor(sq,  off, 32);
    }
    const float mu   = sum * (1.0f / DDIM);
    const float var  = sq * (1.0f / DDIM) - mu * mu;
    const float rsig = rsqrtf(var + 1e-5f);

    const float4* vrow = (const float4*)(video + gbase + (size_t)r * DDIM); // L2 hit
    float4*       orow = (float4*)(out + gbase + (size_t)r * DDIM);
    const float4* g4 = (const float4*)ln_gamma;
    const float4* b4 = (const float4*)ln_beta;
    #pragma unroll
    for (int i = 0; i < 6; ++i) {
      const int c = lane + i * 32;       // float4 index within the row
      float4 vv = vrow[c], gg = g4[c], bb = b4[c];
      float4 o;
      o.x = vv.x + (vbuf[i * 4 + 0] - mu) * rsig * gg.x + bb.x;
      o.y = vv.y + (vbuf[i * 4 + 1] - mu) * rsig * gg.y + bb.y;
      o.z = vv.z + (vbuf[i * 4 + 2] - mu) * rsig * gg.z + bb.z;
      o.w = vv.w + (vbuf[i * 4 + 3] - mu) * rsig * gg.w + bb.w;
      orow[c] = o;
    }
  }
}

extern "C" void kernel_launch(void* const* d_in, const int* in_sizes, int n_in,
                              void* d_out, int out_size, void* d_ws, size_t ws_size,
                              hipStream_t stream) {
  const float* video = (const float*)d_in[0];
  const float* gamma = (const float*)d_in[1];
  const float* beta  = (const float*)d_in[2];
  float* out = (float*)d_out;

  // Allow >64KB dynamic LDS (capture-safe, deterministic, idempotent).
  hipFuncSetAttribute((const void*)msw_attn_ln_kernel,
                      hipFuncAttributeMaxDynamicSharedMemorySize,
                      (int)SMEM_BYTES);

  const int tiles = in_sizes[0] / (TILE_T * DDIM);   // B * T / 64 = 512
  msw_attn_ln_kernel<<<tiles, 256, (unsigned)SMEM_BYTES, stream>>>(
      video, gamma, beta, out);
}
